// GRAM_v2_1503238553830
// MI455X (gfx1250) — matrix operand order repaired and checked by reading, not compiled, not run on hardware
//
#include <hip/hip_runtime.h>
#include <math.h>

typedef __attribute__((ext_vector_type(16))) _Float16 v16h;
typedef __attribute__((ext_vector_type(8)))  _Float16 v8h;
typedef __attribute__((ext_vector_type(2)))  _Float16 h2;
typedef __attribute__((ext_vector_type(8)))  float    v8f;

#define LSEQ 96
#define DDIM 512
#define RDIM 64
#define HDIM 128
#define THRESH 0.3f
#define NEGINF (-3.0e38f)

__device__ __forceinline__ float wr_add(float v) {
#pragma unroll
  for (int off = 16; off > 0; off >>= 1) v += __shfl_xor(v, off, 32);
  return v;
}
__device__ __forceinline__ float wr_max(float v) {
#pragma unroll
  for (int off = 16; off > 0; off >>= 1) v = fmaxf(v, __shfl_xor(v, off, 32));
  return v;
}
__device__ __forceinline__ float wr_min(float v) {
#pragma unroll
  for (int off = 16; off > 0; off >>= 1) v = fminf(v, __shfl_xor(v, off, 32));
  return v;
}

// Blend that lowers to V_BFI_B32: (a & m) | (b & ~m). Bitwise => cannot be
// re-if-converted into branches, and needs no NaN canonicalization.
__device__ __forceinline__ int blendi(int m, int a, int b) {
  return (a & m) | (b & ~m);
}

// Fully bitwise branchless top-3 insert. Three compare masks drive BFI blends
// for both values (as bit patterns) and indices; no fmin/fmax (avoids the
// v_max_num canonicalize traffic clang emits around float min/max).
__device__ __forceinline__ void top3_insert(float s, int m,
                                            float& t0, float& t1, float& t2,
                                            int& i0, int& i1, int& i2) {
  const int g0 = (s > t0) ? -1 : 0;
  const int g1 = (s > t1) ? -1 : 0;
  const int g2 = (s > t2) ? -1 : 0;
  const int sb = __float_as_int(s);
  const int b0 = __float_as_int(t0);
  const int b1 = __float_as_int(t1);
  const int b2 = __float_as_int(t2);
  int n2v = blendi(g1, b1, blendi(g2, sb, b2));  // beats t1 -> t1 demotes; else maybe s
  int n2i = blendi(g1, i1, blendi(g2, m,  i2));
  int n1v = blendi(g0, b0, blendi(g1, sb, b1));  // beats t0 -> t0 demotes; else maybe s
  int n1i = blendi(g0, i0, blendi(g1, m,  i1));
  int n0v = blendi(g0, sb, b0);
  int n0i = blendi(g0, m,  i0);
  t0 = __int_as_float(n0v); t1 = __int_as_float(n1v); t2 = __int_as_float(n2v);
  i0 = n0i; i1 = n1i; i2 = n2i;
}

// ---------------------------------------------------------------------------
// Kernel 1: stream x -> out copy, xf (per-l mean over D), pooled (mean over L),
//           then stats -> encoder linear -> LayerNorm -> l2norm -> q (f16)
// One block per batch row b; 256 threads (8 waves), float2 per thread.
// ---------------------------------------------------------------------------
__global__ void __launch_bounds__(256)
k_encode(const float* __restrict__ x,
         const float* __restrict__ enc_w, const float* __restrict__ enc_b,
         const float* __restrict__ ln_g,  const float* __restrict__ ln_b,
         float* __restrict__ out_x, float* __restrict__ pooled,
         _Float16* __restrict__ qh)
{
  const int b = blockIdx.x;
  const int t = threadIdx.x;
  const float2* __restrict__ xb = (const float2*)(x + (size_t)b * LSEQ * DDIM);
  float2* __restrict__ ob = (float2*)(out_x + (size_t)b * LSEQ * DDIM);

  __shared__ float xf[LSEQ];
  if (t < LSEQ) xf[t] = 0.f;
  __syncthreads();

  float px = 0.f, py = 0.f;
  for (int l = 0; l < LSEQ; ++l) {
    float2 v = xb[l * (DDIM / 2) + t];
    ob[l * (DDIM / 2) + t] = v;
    px += v.x; py += v.y;
    float s = wr_add(v.x + v.y);
    if ((t & 31) == 0) atomicAdd(&xf[l], s);
  }
  pooled[(size_t)b * DDIM + 2 * t]     = px * (1.f / LSEQ);
  pooled[(size_t)b * DDIM + 2 * t + 1] = py * (1.f / LSEQ);
  __syncthreads();

  if (t < 32) {
    float a0 = xf[t]      * (1.f / DDIM);
    float a1 = xf[t + 32] * (1.f / DDIM);
    float a2 = xf[t + 64] * (1.f / DDIM);
    float sum = wr_add(a0 + a1 + a2);
    float ssq = wr_add(a0 * a0 + a1 * a1 + a2 * a2);
    float mx  = wr_max(fmaxf(a0, fmaxf(a1, a2)));
    float mn  = wr_min(fminf(a0, fminf(a1, a2)));
    float mean = sum * (1.f / LSEQ);
    float var  = (ssq - (float)LSEQ * mean * mean) * (1.f / (LSEQ - 1));  // unbiased
    float stdv = sqrtf(fmaxf(var, 0.f));
    float trend = (xf[LSEQ - 1] - xf[0]) * (1.f / DDIM);

    const float st0 = mean, st1 = stdv, st2 = mx, st3 = mn, st4 = trend;
    const int j0 = t, j1 = t + 32;
    float h0 = enc_b[j0] + st0 * enc_w[j0 * 5 + 0] + st1 * enc_w[j0 * 5 + 1] +
               st2 * enc_w[j0 * 5 + 2] + st3 * enc_w[j0 * 5 + 3] + st4 * enc_w[j0 * 5 + 4];
    float h1 = enc_b[j1] + st0 * enc_w[j1 * 5 + 0] + st1 * enc_w[j1 * 5 + 1] +
               st2 * enc_w[j1 * 5 + 2] + st3 * enc_w[j1 * 5 + 3] + st4 * enc_w[j1 * 5 + 4];
    float hs  = wr_add(h0 + h1);
    float hss = wr_add(h0 * h0 + h1 * h1);
    float mu  = hs * (1.f / RDIM);
    float vr  = hss * (1.f / RDIM) - mu * mu;
    float inv = rsqrtf(vr + 1e-5f);
    float q0 = (h0 - mu) * inv * ln_g[j0] + ln_b[j0];
    float q1 = (h1 - mu) * inv * ln_g[j1] + ln_b[j1];
    float nrm = sqrtf(wr_add(q0 * q0 + q1 * q1));
    float sc = 1.f / fmaxf(nrm, 1e-12f);
    qh[(size_t)b * RDIM + j0] = (_Float16)(q0 * sc);
    qh[(size_t)b * RDIM + j1] = (_Float16)(q1 * sc);
  }
}

// ---------------------------------------------------------------------------
// Kernel 2: L2-normalize memory_keys rows -> f16. One wave per row.
// ---------------------------------------------------------------------------
__global__ void __launch_bounds__(256)
k_keys(const float* __restrict__ mk, _Float16* __restrict__ knh, int M)
{
  const int lane = threadIdx.x & 31;
  const int wid  = (int)((blockIdx.x * blockDim.x + threadIdx.x) >> 5);
  const int nw   = (int)((gridDim.x * blockDim.x) >> 5);
  for (int m = wid; m < M; m += nw) {
    float2 v = ((const float2*)(mk + (size_t)m * RDIM))[lane];
    float ss = wr_add(v.x * v.x + v.y * v.y);
    float sc = 1.f / fmaxf(sqrtf(ss), 1e-12f);
    h2 o;
    o.x = (_Float16)(v.x * sc);
    o.y = (_Float16)(v.y * sc);
    ((h2*)(knh + (size_t)m * RDIM))[lane] = o;
  }
}

// ---------------------------------------------------------------------------
// Kernel 3: rowsum[d] = sum_r ctx_w[d,r]; zero valid flag (ws is not re-poisoned)
// ---------------------------------------------------------------------------
__global__ void k_prep(const float* __restrict__ ctx_w,
                       float* __restrict__ rowsum, int* __restrict__ valid)
{
  const int t = threadIdx.x;  // 512
  float s = 0.f;
#pragma unroll 8
  for (int r = 0; r < RDIM; ++r) s += ctx_w[t * RDIM + r];
  rowsum[t] = s;
  if (t == 0) *valid = 0;
}

// ---------------------------------------------------------------------------
// Kernel 4: fused sim GEMM (f16 WMMA, f32 accum) + streaming branchless top-3.
// ---------------------------------------------------------------------------
#define NW 8

__device__ __forceinline__ void sim_tile_body(const _Float16* __restrict__ krow,
                                              int midx, int inbm,
                                              const v16h& A0, const v16h& A1,
                                              float* tv0, float* tv1, float* tv2,
                                              int* ti0, int* ti1, int* ti2)
{
  // B (32x16 f16): lane holds column m, K = 16*hi + 32*chunk .. +15 (contiguous)
  v8h b00 = *(const v8h*)(krow);
  v8h b01 = *(const v8h*)(krow + 8);
  v8h b10 = *(const v8h*)(krow + 32);
  v8h b11 = *(const v8h*)(krow + 40);
  v16h B0 = __builtin_shufflevector(b00, b01, 0,1,2,3,4,5,6,7,8,9,10,11,12,13,14,15);
  v16h B1 = __builtin_shufflevector(b10, b11, 0,1,2,3,4,5,6,7,8,9,10,11,12,13,14,15);

  v8f acc = {0.f, 0.f, 0.f, 0.f, 0.f, 0.f, 0.f, 0.f};
  acc = __builtin_amdgcn_wmma_f32_16x16x32_f16(false, A0, false, B0, (short)0, acc, false, false);
  acc = __builtin_amdgcn_wmma_f32_16x16x32_f16(false, A1, false, B1, (short)0, acc, false, false);

#pragma unroll
  for (int v = 0; v < 8; ++v) {
    // OOB folds into the value bit-pattern (bitwise; inbm == -1 in hot loop and
    // the blend folds away at compile time there)
    float s = __int_as_float(blendi(inbm, __float_as_int(acc[v]), __float_as_int(NEGINF)));
    top3_insert(s, midx, tv0[v], tv1[v], tv2[v], ti0[v], ti1[v], ti2[v]);
  }
}

__global__ void __launch_bounds__(32 * NW)
k_sim_topk(const _Float16* __restrict__ qh, const _Float16* __restrict__ knh,
           float* __restrict__ topv, int* __restrict__ topi,
           int* __restrict__ valid, int M)
{
  const int lane = threadIdx.x & 31;
  const int wave = threadIdx.x >> 5;
  const int col  = lane & 15;
  const int hi   = lane >> 4;
  const int bbase = blockIdx.x * 16;
  const int mtiles = (M + 15) >> 4;
  const int mfull  = M >> 4;             // tiles with no OOB columns

  // A (16x32 f16): lane<16 -> K {0..15}, lane>=16 -> K {16..31} (contiguous)
  const _Float16* qrow = qh + (size_t)(bbase + col) * RDIM + hi * 16;
  v8h a00 = *(const v8h*)(qrow);
  v8h a01 = *(const v8h*)(qrow + 8);
  v8h a10 = *(const v8h*)(qrow + 32);
  v8h a11 = *(const v8h*)(qrow + 40);
  v16h A0 = __builtin_shufflevector(a00, a01, 0,1,2,3,4,5,6,7,8,9,10,11,12,13,14,15);
  v16h A1 = __builtin_shufflevector(a10, a11, 0,1,2,3,4,5,6,7,8,9,10,11,12,13,14,15);

  float tv0[8], tv1[8], tv2[8];
  int   ti0[8], ti1[8], ti2[8];
#pragma unroll
  for (int v = 0; v < 8; ++v) { tv0[v] = tv1[v] = tv2[v] = NEGINF; ti0[v] = ti1[v] = ti2[v] = 0; }

  // Strength-reduced hot loop: running pointer + running m index, no per-iter
  // 64-bit shift/re-derivation from the loop counter.
  const int step = NW * 16;                                // m advance per iter
  const size_t pstep = (size_t)step * RDIM;                // halves per iter
  int midx = (wave << 4) + col;
  const _Float16* krow = knh + (size_t)midx * RDIM + hi * 16;
  int nfull = (mfull > wave) ? ((mfull - wave + NW - 1) / NW) : 0;
  for (int it = 0; it < nfull; ++it) {
    sim_tile_body(krow, midx, -1, A0, A1, tv0, tv1, tv2, ti0, ti1, ti2);
    krow += pstep;
    midx += step;
  }
  // Tail tile (empty when M % 16 == 0)
  for (int mt = wave + nfull * NW; mt < mtiles; mt += NW) {
    int mr = (mt << 4) + col;
    int mi = mr;
    if (mr >= M) mr = M - 1;
    const _Float16* kr = knh + (size_t)mr * RDIM + hi * 16;
    sim_tile_body(kr, mi, (mi < M) ? -1 : 0, A0, A1, tv0, tv1, tv2, ti0, ti1, ti2);
  }

  __shared__ float lv[NW * 32 * 8 * 3];
  __shared__ int   li[NW * 32 * 8 * 3];
  const int base = ((wave * 32 + lane) * 8) * 3;
#pragma unroll
  for (int v = 0; v < 8; ++v) {
    lv[base + v * 3 + 0] = tv0[v]; li[base + v * 3 + 0] = ti0[v];
    lv[base + v * 3 + 1] = tv1[v]; li[base + v * 3 + 1] = ti1[v];
    lv[base + v * 3 + 2] = tv2[v]; li[base + v * 3 + 2] = ti2[v];
  }
  __syncthreads();

  if (threadIdx.x < 16) {
    const int r = threadIdx.x;
    const int lbase = (r < 8) ? 0 : 16;   // lanes holding this row
    const int v = r & 7;                  // vgpr slot for this row
    float b0 = NEGINF, b1 = NEGINF, b2 = NEGINF;
    int j0 = 0, j1 = 0, j2 = 0;
    for (int w = 0; w < NW; ++w)
      for (int ln = 0; ln < 16; ++ln) {
        const int idx = ((w * 32 + lbase + ln) * 8 + v) * 3;
#pragma unroll
        for (int s3 = 0; s3 < 3; ++s3)
          top3_insert(lv[idx + s3], li[idx + s3], b0, b1, b2, j0, j1, j2);
      }
    const int b = bbase + r;
    topv[b * 3 + 0] = b0; topv[b * 3 + 1] = b1; topv[b * 3 + 2] = b2;
    topi[b * 3 + 0] = j0; topi[b * 3 + 1] = j1; topi[b * 3 + 2] = j2;
    if (b0 > THRESH) atomicOr(valid, 1);
  }
}

// ---------------------------------------------------------------------------
// Kernel 5: gating MLP (exact GELU) + sigmoid gate + softmax-3 feat + ctx.
// Block = 16 b-rows, 128 threads (one per hidden unit j).
// ---------------------------------------------------------------------------
__global__ void __launch_bounds__(128)
k_gate_ctx(const float* __restrict__ pooled,
           const float* __restrict__ topv, const int* __restrict__ topi,
           const float* __restrict__ mv,
           const float* __restrict__ gw1, const float* __restrict__ gb1,
           const float* __restrict__ gw2, const float* __restrict__ gb2,
           const float* __restrict__ rowsum, const float* __restrict__ ctx_b,
           const int* __restrict__ valid,
           float* __restrict__ out_ctx, float* __restrict__ out_gate)
{
  const int bb = blockIdx.x * 16;
  const int t = threadIdx.x;

  __shared__ float pl[16][DDIM];
  __shared__ float hl[16][HDIM];
  __shared__ float feat[16];
  __shared__ float msim[16];

  for (int i = t; i < 16 * DDIM; i += 128)
    pl[i >> 9][i & 511] = pooled[(size_t)(bb + (i >> 9)) * DDIM + (i & 511)];

  if (t < 16) {
    const int b = bb + t;
    float s0 = topv[b * 3 + 0], s1 = topv[b * 3 + 1], s2 = topv[b * 3 + 2];
    msim[t] = s0;  // sorted descending -> max
    float e0 = 1.f, e1 = expf(s1 - s0), e2 = expf(s2 - s0);
    float den = e0 + e1 + e2;
    float r0 = mv[topi[b * 3 + 0]], r1 = mv[topi[b * 3 + 1]], r2 = mv[topi[b * 3 + 2]];
    feat[t] = (r0 * e0 + r1 * e1 + r2 * e2) / den;
  }
  __syncthreads();

  float acc[16];
#pragma unroll
  for (int b = 0; b < 16; ++b) acc[b] = gb1[t];
  for (int d = 0; d < DDIM; ++d) {
    float w = gw1[t * (DDIM + 1) + d];
#pragma unroll
    for (int b = 0; b < 16; ++b) acc[b] = fmaf(w, pl[b][d], acc[b]);
  }
  const float wlast = gw1[t * (DDIM + 1) + DDIM];
  const float w2 = gw2[t];
#pragma unroll
  for (int b = 0; b < 16; ++b) {
    float hx = acc[b] + wlast * msim[b];
    float g = 0.5f * hx * (1.f + erff(hx * 0.7071067811865475f));  // exact GELU
    hl[b][t] = g * w2;
  }
  __syncthreads();

  const bool ok = (*valid) != 0;
  if (t < 16) {
    float s = 0.f;
    for (int j = 0; j < HDIM; ++j) s += hl[t][j];
    float gate = 1.f / (1.f + expf(-(s + gb2[0])));
    out_gate[bb + t] = ok ? gate : 1.f;
  }
  for (int i = t; i < 16 * DDIM; i += 128) {
    const int b = i >> 9, d = i & 511;
    float c = ok ? fmaf(feat[b], rowsum[d], ctx_b[d]) : 0.f;
    out_ctx[(size_t)(bb + b) * DDIM + d] = c;
  }
}

// ---------------------------------------------------------------------------
extern "C" void kernel_launch(void* const* d_in, const int* in_sizes, int n_in,
                              void* d_out, int out_size, void* d_ws, size_t ws_size,
                              hipStream_t stream)
{
  const float* x     = (const float*)d_in[0];
  const float* mk    = (const float*)d_in[1];
  const float* mv    = (const float*)d_in[2];
  const float* enc_w = (const float*)d_in[3];
  const float* enc_b = (const float*)d_in[4];
  const float* ln_g  = (const float*)d_in[5];
  const float* ln_b  = (const float*)d_in[6];
  const float* gw1   = (const float*)d_in[7];
  const float* gb1   = (const float*)d_in[8];
  const float* gw2   = (const float*)d_in[9];
  const float* gb2   = (const float*)d_in[10];
  const float* ctxw  = (const float*)d_in[11];
  const float* ctxb  = (const float*)d_in[12];

  const int B = in_sizes[0] / (LSEQ * DDIM);
  const int M = in_sizes[1] / RDIM;

  char* wsp = (char*)d_ws;
  size_t off = 0;
  auto alloc = [&](size_t bytes) -> char* {
    char* p = wsp + off;
    off += (bytes + 255) & ~(size_t)255;
    return p;
  };
  _Float16* qh  = (_Float16*)alloc((size_t)B * RDIM * sizeof(_Float16));
  _Float16* knh = (_Float16*)alloc((size_t)M * RDIM * sizeof(_Float16));
  float* pooled = (float*)alloc((size_t)B * DDIM * sizeof(float));
  float* topv   = (float*)alloc((size_t)B * 3 * sizeof(float));
  int*   topi   = (int*)alloc((size_t)B * 3 * sizeof(int));
  float* rowsum = (float*)alloc(DDIM * sizeof(float));
  int*   valid  = (int*)alloc(256);

  float* out_x    = (float*)d_out;
  float* out_ctx  = out_x + (size_t)B * LSEQ * DDIM;
  float* out_gate = out_ctx + (size_t)B * DDIM;

  k_encode<<<B, 256, 0, stream>>>(x, enc_w, enc_b, ln_g, ln_b, out_x, pooled, qh);
  k_keys<<<256, 256, 0, stream>>>(mk, knh, M);
  k_prep<<<1, DDIM, 0, stream>>>(ctxw, rowsum, valid);
  k_sim_topk<<<B / 16, 32 * NW, 0, stream>>>(qh, knh, topv, topi, valid, M);
  k_gate_ctx<<<B / 16, 128, 0, stream>>>(pooled, topv, topi, mv, gw1, gb1, gw2, gb2,
                                         rowsum, ctxb, valid, out_ctx, out_gate);
}
